// Attention_41137196761104
// MI455X (gfx1250) — compile-verified, matrix operand order checked by reading, not run on hardware
//
#include <hip/hip_runtime.h>
#include <hip/hip_bf16.h>

// Problem constants
#define BATCH 2
#define SEQ   2048
#define DMODEL 2048
#define NHEAD 16
#define HDIM  128
#define PREFIX 512
#define STOT  (PREFIX + SEQ)          // 2560
#define M_ROWS (BATCH * SEQ)          // 4096
#define QKV_N (3 * DMODEL)            // 6144
#define OUT1_ELEMS (BATCH * SEQ * DMODEL)        // 8,388,608
#define KV_HALF (BATCH * NHEAD * STOT * HDIM)    // 10,485,760 (one of k/v)

typedef __attribute__((ext_vector_type(16))) _Float16 v16h;
typedef __attribute__((ext_vector_type(8)))  float    v8f;

// Fragment union: 16 halves = two 16-byte quads (matches ISA 7.12.2 layout:
// element e -> K = (e&7) + 8*half + (e>=8 ? 16 : 0), i.e. two contiguous
// 8-half runs at +8*half and +16+8*half within a lane's row).
union AF { v16h v; _Float16 h[16]; float4 q[2]; };
union F8 { v8f  v; float    f[8];  };
union H8 { float4 q; _Float16 h[8]; };

__device__ __forceinline__ v8f wmma_f16(v16h a, v16h b, v8f c) {
  return __builtin_amdgcn_wmma_f32_16x16x32_f16(false, a, false, b, (short)0, c,
                                                false, false);
}

// Load a 16-half fragment from a row pointer (16B-aligned base, k0 mult of 32).
__device__ __forceinline__ void load_frag(AF& f, const _Float16* p, int half) {
  f.q[0] = *(const float4*)(p + 8 * half);
  f.q[1] = *(const float4*)(p + 16 + 8 * half);
}

// ---------------------------------------------------------------------------
// Prep 1: elementwise f32 -> f16 (x -> Xh).  8 elems/thread, b128 in/out.
// ---------------------------------------------------------------------------
__global__ void cvt_f16(const float* __restrict__ in, _Float16* __restrict__ out) {
  const int i = (blockIdx.x * 256 + threadIdx.x) * 8;
  const float4 a = *(const float4*)(in + i);
  const float4 b = *(const float4*)(in + i + 4);
  H8 t;
  t.h[0] = (_Float16)a.x; t.h[1] = (_Float16)a.y;
  t.h[2] = (_Float16)a.z; t.h[3] = (_Float16)a.w;
  t.h[4] = (_Float16)b.x; t.h[5] = (_Float16)b.y;
  t.h[6] = (_Float16)b.z; t.h[7] = (_Float16)b.w;
  *(float4*)(out + i) = t.q;
}

// ---------------------------------------------------------------------------
// Prep 2: weight transpose + convert.  in: f32 (K x N) -> out: f16 (N x K).
// 32x32 tile through LDS; coalesced reads (n) and writes (k).
// ---------------------------------------------------------------------------
__global__ void transpose_cvt(const float* __restrict__ in,
                              _Float16* __restrict__ out, int K, int N) {
  __shared__ float tile[32][33];
  const int n0 = blockIdx.x * 32, k0 = blockIdx.y * 32;
  const int tx = threadIdx.x, ty = threadIdx.y;   // 32 x 8
#pragma unroll
  for (int i = 0; i < 4; ++i)
    tile[ty + 8 * i][tx] = in[(k0 + ty + 8 * i) * N + n0 + tx];   // [k][n]
  __syncthreads();
#pragma unroll
  for (int i = 0; i < 4; ++i)
    out[(n0 + ty + 8 * i) * K + k0 + tx] = (_Float16)tile[tx][ty + 8 * i];
}

// ---------------------------------------------------------------------------
// Kernel 1: QKV projection.  Xh (4096x2048 f16) @ Wqkvt^T (f16, stored 6144x2048)
// + bqkv.  q -> Qr f16 (B,H,N,HD) (roped later, in place);
// k,v -> kvout f32 (d_out KV region, pre-RoPE, required output).
// Wave: 32x64 tile; block 4 waves = 64x128.
// ---------------------------------------------------------------------------
__global__ void qkv_gemm(const _Float16* __restrict__ A,
                         const _Float16* __restrict__ Bt,
                         const float* __restrict__ bias,
                         _Float16* __restrict__ Qr, float* __restrict__ kvout) {
  const int lane = threadIdx.x & 31;
  const int wid  = threadIdx.x >> 5;
  const int m0 = blockIdx.x * 64  + (wid >> 1) * 32;
  const int n0 = blockIdx.y * 128 + (wid & 1) * 64;
  const int row  = lane & 15;
  const int half = lane >> 4;

  F8 acc[2][4];
  for (int t = 0; t < 2; ++t)
    for (int j = 0; j < 4; ++j)
      for (int r = 0; r < 8; ++r) acc[t][j].f[r] = 0.0f;

  for (int k0 = 0; k0 < DMODEL; k0 += 32) {
    __builtin_prefetch(A + (m0 + row) * DMODEL + k0 + 256, 0, 1);
    AF a[2], bf[4];
#pragma unroll
    for (int t = 0; t < 2; ++t)
      load_frag(a[t], A + (m0 + t * 16 + row) * DMODEL + k0, half);
#pragma unroll
    for (int j = 0; j < 4; ++j)
      load_frag(bf[j], Bt + (n0 + j * 16 + row) * DMODEL + k0, half);
#pragma unroll
    for (int t = 0; t < 2; ++t)
#pragma unroll
      for (int j = 0; j < 4; ++j)
        acc[t][j].v = wmma_f16(a[t].v, bf[j].v, acc[t][j].v);
  }

  for (int t = 0; t < 2; ++t)
    for (int j = 0; j < 4; ++j)
      for (int r = 0; r < 8; ++r) {
        const int m = m0 + t * 16 + r + 8 * half;
        const int n = n0 + j * 16 + row;
        const float val = acc[t][j].f[r] + bias[n];
        const int sel = n >> 11;           // 0=q 1=k 2=v
        const int rem = n & 2047;
        const int h = rem >> 7, hd = rem & 127;
        const int bb = m >> 11, i = m & 2047;
        const int bh = bb * NHEAD + h;
        if (sel == 0) {
          Qr[(bh * SEQ + i) * HDIM + hd] = (_Float16)val;
        } else {
          int off = (bh * STOT + PREFIX + i) * HDIM + hd;
          if (sel == 2) off += KV_HALF;
          kvout[off] = val;
        }
      }
}

// ---------------------------------------------------------------------------
// Kernel 2: prefix copy + RoPE.  Ropes Qr in place (pre-scaled by 1/sqrt(HD)),
// produces roped f16 K (row-major) and f16 V transposed (B,H,HD,STOT);
// fills prefix region of d_out KV.  One 64-thread block per (b,h,s) row.
// ---------------------------------------------------------------------------
__global__ void rope_assemble(const float* __restrict__ prefix_k,
                              const float* __restrict__ prefix_v,
                              float* __restrict__ kvout,
                              _Float16* __restrict__ Qr,
                              _Float16* __restrict__ Kr,
                              _Float16* __restrict__ Vt) {
  const int rid = blockIdx.x;        // 0 .. B*H*STOT-1
  const int j = threadIdx.x;         // 0..63
  const int bh = rid / STOT;
  const int s = rid % STOT;
  const int base = (bh * STOT + s) * HDIM;

  float k1, k2, v1, v2;
  if (s < PREFIX) {
    const int pb = (bh * PREFIX + s) * HDIM;
    k1 = prefix_k[pb + j];      k2 = prefix_k[pb + 64 + j];
    v1 = prefix_v[pb + j];      v2 = prefix_v[pb + 64 + j];
    kvout[base + j] = k1;               kvout[base + 64 + j] = k2;
    kvout[KV_HALF + base + j] = v1;     kvout[KV_HALF + base + 64 + j] = v2;
  } else {
    k1 = kvout[base + j];               k2 = kvout[base + 64 + j];
    v1 = kvout[KV_HALF + base + j];     v2 = kvout[KV_HALF + base + 64 + j];
  }

  // inv_freq = 10000^(-j/64) = exp(-j * ln(10000)/64)
  const float inv = __expf(-0.14391565298f * (float)j);
  float sn, cs;
  __sincosf((float)s * inv, &sn, &cs);

  Kr[base + j]      = (_Float16)(k1 * cs - k2 * sn);
  Kr[base + 64 + j] = (_Float16)(k2 * cs + k1 * sn);
  // V transposed: (bh, hd, s)
  Vt[(bh * HDIM + j) * STOT + s]      = (_Float16)v1;
  Vt[(bh * HDIM + 64 + j) * STOT + s] = (_Float16)v2;

  if (s < SEQ) {
    const int qb = (bh * SEQ + s) * HDIM;
    const float q1 = (float)Qr[qb + j], q2 = (float)Qr[qb + 64 + j];
    const float qs = 0.08838834764831845f;  // 1/sqrt(128), folded into Q
    Qr[qb + j]      = (_Float16)((q1 * cs - q2 * sn) * qs);
    Qr[qb + 64 + j] = (_Float16)((q2 * cs + q1 * sn) * qs);
  }
}

// ---------------------------------------------------------------------------
// Kernel 3: flash attention.  One wave owns 16 query rows; KV chunks of 32.
// Scores: 8 wmma; online softmax (shfl reductions); P transposed through
// per-wave f16 LDS tile (b128 ds loads); P@V: 8 wmma with transposed-V frags.
// ---------------------------------------------------------------------------
__global__ void flash_attn(const _Float16* __restrict__ Qr,
                           const _Float16* __restrict__ Kr,
                           const _Float16* __restrict__ Vt,
                           _Float16* __restrict__ AO) {
  __shared__ _Float16 plds[4][16 * 32];
  const int lane = threadIdx.x & 31;
  const int w = threadIdx.x >> 5;
  const int wid = blockIdx.x * 4 + w;
  const int qt = wid & 127;          // q tile within (b,h)
  const int bh = wid >> 7;           // 0..31
  const int bb = bh >> 4, h = bh & 15;
  const int qb = qt * 16;
  const int row = lane & 15, half = lane >> 4;

  const _Float16* Qp = Qr + (bh * SEQ + qb) * HDIM;
  const _Float16* Kp = Kr + bh * STOT * HDIM;
  const _Float16* Vp = Vt + bh * HDIM * STOT;

  AF qf[4];
#pragma unroll
  for (int c = 0; c < 4; ++c)
    load_frag(qf[c], Qp + row * HDIM + c * 32, half);

  F8 acc[8];
  for (int j = 0; j < 8; ++j)
    for (int r = 0; r < 8; ++r) acc[j].f[r] = 0.0f;
  float mrow[8], lrow[8];
  for (int r = 0; r < 8; ++r) { mrow[r] = -3.0e38f; lrow[r] = 0.0f; }

  const int ncols = qb + 16 + PREFIX;           // exclusive valid-col bound
  const int nch = (ncols + 31) >> 5;
  for (int ch = 0; ch < nch; ++ch) {
    const int c0 = ch * 32;
    F8 s0, s1;
    for (int r = 0; r < 8; ++r) { s0.f[r] = 0.0f; s1.f[r] = 0.0f; }
#pragma unroll
    for (int ks = 0; ks < 4; ++ks) {
      AF b0, b1;
      load_frag(b0, Kp + (c0 + row) * HDIM + ks * 32, half);
      load_frag(b1, Kp + (c0 + 16 + row) * HDIM + ks * 32, half);
      s0.v = wmma_f16(qf[ks].v, b0.v, s0.v);
      s1.v = wmma_f16(qf[ks].v, b1.v, s1.v);
    }
    // Online softmax update; write exp(P) (f16) into this wave's LDS tile.
#pragma unroll
    for (int r = 0; r < 8; ++r) {
      const int qg = qb + r + 8 * half;
      float x0 = (c0 + row <= qg + PREFIX) ? s0.f[r] : -3.0e38f;
      float x1 = (c0 + 16 + row <= qg + PREFIX) ? s1.f[r] : -3.0e38f;
      float mx = fmaxf(x0, x1);
      for (int d = 1; d < 16; d <<= 1) mx = fmaxf(mx, __shfl_xor(mx, d));
      const float mnew = fmaxf(mrow[r], mx);
      const float scale = __expf(mrow[r] - mnew);
      const float p0 = __expf(x0 - mnew);
      const float p1 = __expf(x1 - mnew);
      mrow[r] = mnew;
      float ps = p0 + p1;
      for (int d = 1; d < 16; d <<= 1) ps += __shfl_xor(ps, d);
      lrow[r] = lrow[r] * scale + ps;
#pragma unroll
      for (int j = 0; j < 8; ++j) acc[j].f[r] *= scale;
      plds[w][(r + 8 * half) * 32 + row] = (_Float16)p0;
      plds[w][(r + 8 * half) * 32 + 16 + row] = (_Float16)p1;
    }
    asm volatile("s_wait_dscnt 0" ::: "memory");   // intra-wave LDS RAW fence
    AF pa;
    pa.q[0] = *(const float4*)&plds[w][row * 32 + 8 * half];
    pa.q[1] = *(const float4*)&plds[w][row * 32 + 16 + 8 * half];
#pragma unroll
    for (int j = 0; j < 8; ++j) {
      AF bv;   // V^T row: lane = hd column, elements = contiguous s run
      load_frag(bv, Vp + (j * 16 + row) * STOT + c0, half);
      acc[j].v = wmma_f16(pa.v, bv.v, acc[j].v);
    }
  }

  // Normalize and store (B, N, D) with D index = h*128 + hd.
  for (int j = 0; j < 8; ++j)
    for (int r = 0; r < 8; ++r) {
      const int m = r + 8 * half;
      const float o = acc[j].f[r] / lrow[r];
      AO[(bb * SEQ + qb + m) * DMODEL + h * HDIM + j * 16 + row] = (_Float16)o;
    }
}

// ---------------------------------------------------------------------------
// Kernel 4: FF projection.  AO (4096x2048 f16) @ Wfft^T (f16, 2048x2048) + bff
// -> d_out part 1 (f32).
// ---------------------------------------------------------------------------
__global__ void ff_gemm(const _Float16* __restrict__ A,
                        const _Float16* __restrict__ Bt,
                        const float* __restrict__ bias, float* __restrict__ out) {
  const int lane = threadIdx.x & 31;
  const int wid  = threadIdx.x >> 5;
  const int m0 = blockIdx.x * 64  + (wid >> 1) * 32;
  const int n0 = blockIdx.y * 128 + (wid & 1) * 64;
  const int row  = lane & 15;
  const int half = lane >> 4;

  F8 acc[2][4];
  for (int t = 0; t < 2; ++t)
    for (int j = 0; j < 4; ++j)
      for (int r = 0; r < 8; ++r) acc[t][j].f[r] = 0.0f;

  for (int k0 = 0; k0 < DMODEL; k0 += 32) {
    __builtin_prefetch(Bt + (n0 + row) * DMODEL + k0 + 256, 0, 1);
    AF a[2], bf[4];
#pragma unroll
    for (int t = 0; t < 2; ++t)
      load_frag(a[t], A + (m0 + t * 16 + row) * DMODEL + k0, half);
#pragma unroll
    for (int j = 0; j < 4; ++j)
      load_frag(bf[j], Bt + (n0 + j * 16 + row) * DMODEL + k0, half);
#pragma unroll
    for (int t = 0; t < 2; ++t)
#pragma unroll
      for (int j = 0; j < 4; ++j)
        acc[t][j].v = wmma_f16(a[t].v, bf[j].v, acc[t][j].v);
  }

  for (int t = 0; t < 2; ++t)
    for (int j = 0; j < 4; ++j)
      for (int r = 0; r < 8; ++r) {
        const int m = m0 + t * 16 + r + 8 * half;
        const int n = n0 + j * 16 + row;
        out[m * DMODEL + n] = acc[t][j].f[r] + bias[n];
      }
}

// ---------------------------------------------------------------------------
extern "C" void kernel_launch(void* const* d_in, const int* in_sizes, int n_in,
                              void* d_out, int out_size, void* d_ws, size_t ws_size,
                              hipStream_t stream) {
  const float* x        = (const float*)d_in[0];
  // d_in[1] = mask (int32) — causal+prefix mask computed analytically.
  const float* prefix_k = (const float*)d_in[2];
  const float* prefix_v = (const float*)d_in[3];
  const float* Wqkv     = (const float*)d_in[4];
  const float* bqkv     = (const float*)d_in[5];
  const float* Wff      = (const float*)d_in[6];
  const float* bff      = (const float*)d_in[7];

  float* out1  = (float*)d_out;
  float* kvout = out1 + OUT1_ELEMS;     // (2,B,H,STOT,HD) region

  char* ws = (char*)d_ws;
  _Float16* Xh    = (_Float16*)(ws);                          // 16.78 MB
  _Float16* Wqkvt = (_Float16*)(ws + 16777216);               // 25.17 MB
  _Float16* Wfft  = (_Float16*)(ws + 41943040);               //  8.39 MB
  _Float16* Qr    = (_Float16*)(ws + 50331648);               // 16.78 MB
  _Float16* Kr    = (_Float16*)(ws + 67108864);               // 20.97 MB
  _Float16* Vt    = (_Float16*)(ws + 88080384);               // 20.97 MB
  _Float16* AO    = (_Float16*)(ws + 109051904);              // 16.78 MB

  cvt_f16<<<dim3(OUT1_ELEMS / (256 * 8)), 256, 0, stream>>>(x, Xh);
  transpose_cvt<<<dim3(QKV_N / 32, DMODEL / 32), dim3(32, 8), 0, stream>>>(
      Wqkv, Wqkvt, DMODEL, QKV_N);
  transpose_cvt<<<dim3(DMODEL / 32, DMODEL / 32), dim3(32, 8), 0, stream>>>(
      Wff, Wfft, DMODEL, DMODEL);

  qkv_gemm<<<dim3(M_ROWS / 64, QKV_N / 128), 128, 0, stream>>>(Xh, Wqkvt, bqkv,
                                                               Qr, kvout);
  rope_assemble<<<dim3(BATCH * NHEAD * STOT), 64, 0, stream>>>(
      prefix_k, prefix_v, kvout, Qr, Kr, Vt);
  flash_attn<<<dim3(BATCH * NHEAD * (SEQ / 16) / 4), 128, 0, stream>>>(Qr, Kr, Vt,
                                                                       AO);
  ff_gemm<<<dim3(M_ROWS / 64, DMODEL / 128), 128, 0, stream>>>(AO, Wfft, bff,
                                                               out1);
}
